// MultiHeadAttention_18193481466323
// MI455X (gfx1250) — matrix operand order repaired and checked by reading, not compiled, not run on hardware
//
#include <hip/hip_runtime.h>

// ---------------------------------------------------------------------------
// MHA forward for B=2, S=2048, D=1024, H=16, DK=64 on gfx1250 (CDNA5).
// All matmuls use v_wmma_f32_16x16x32_bf16 (wave32 WMMA), f32 accumulation.
// d_out = [out: B*S*D f32][attn: B*H*S*S f32]
// d_ws  = [Q bf16 4096x1024][K bf16][V bf16][X bf16]  (32 MB)
// ---------------------------------------------------------------------------

constexpr int Bc  = 2;
constexpr int Sc  = 2048;
constexpr int Dc  = 1024;
constexpr int Hc  = 16;
constexpr int DKc = 64;

typedef __attribute__((ext_vector_type(16))) __bf16 bf16x16;
typedef __attribute__((ext_vector_type(8)))  __bf16 bf16x8;
typedef __attribute__((ext_vector_type(8)))  float  f32x8;
typedef __attribute__((ext_vector_type(4)))  float  f32x4;

__device__ __forceinline__ __bf16 to_bf16(float f) { return (__bf16)f; }
__device__ __forceinline__ __bf16 to_bf16(__bf16 b) { return b; }

__device__ __forceinline__ void store_out(float* p, float v)  { *p = v; }
__device__ __forceinline__ void store_out(__bf16* p, float v) { *p = to_bf16(v); }

// Convert 16 contiguous f32 -> two bf16x8
__device__ __forceinline__ void cvt16(const float* __restrict__ src,
                                      bf16x8& lo, bf16x8& hi) {
    const f32x4* s4 = (const f32x4*)src;
    f32x4 a = s4[0], b = s4[1], c = s4[2], d = s4[3];
#pragma unroll
    for (int j = 0; j < 4; ++j) {
        lo[j]     = (__bf16)a[j];
        lo[j + 4] = (__bf16)b[j];
        hi[j]     = (__bf16)c[j];
        hi[j + 4] = (__bf16)d[j];
    }
}

// Stage a 16-element contiguous chunk into LDS (b128 stores).
__device__ __forceinline__ void stage_chunk16(const float* __restrict__ src,
                                              __bf16* __restrict__ dst) {
    bf16x8 lo, hi;
    cvt16(src, lo, hi);
    ((bf16x8*)dst)[0] = lo;
    ((bf16x8*)dst)[1] = hi;
}
__device__ __forceinline__ void stage_chunk16(const __bf16* __restrict__ src,
                                              __bf16* __restrict__ dst) {
    ((bf16x8*)dst)[0] = ((const bf16x8*)src)[0];
    ((bf16x8*)dst)[1] = ((const bf16x8*)src)[1];
}

// A fragment: 16x32 bf16, lane m = lane&15, elements 0..7 -> K = half*8 + i,
// elements 8..15 -> K = 16 + half*8 + (i-8).  LDS tile [rows][ldk] bf16,
// sub-tile K offset ko.
__device__ __forceinline__ bf16x16 load_a_frag(const __bf16* lA, int ldk,
                                               int m0, int ko, int lane) {
    int m    = m0 + (lane & 15);
    int half = (lane >> 4) & 1;
    const __bf16* row = lA + m * ldk + ko;
    union { bf16x16 v16; bf16x8 v8[2]; } u;
    u.v8[0] = *(const bf16x8*)(row + half * 8);
    u.v8[1] = *(const bf16x8*)(row + 16 + half * 8);
    return u.v16;
}

// B fragment: 32x16 bf16, lane n = lane&15, elements 0..7 -> K = half*8 + i,
// elements 8..15 -> K = 16 + half*8 + (i-8).
// LDS tile stores B transposed: lB[n][k], [cols][ldk] bf16, K offset ko.
__device__ __forceinline__ bf16x16 load_b_frag(const __bf16* lB, int ldk,
                                               int n0, int ko, int lane) {
    int n  = n0 + (lane & 15);
    int k0 = ko + ((lane >> 4) & 1) * 8;
    const __bf16* row = lB + n * ldk;
    union { bf16x16 v16; bf16x8 v8[2]; } u;
    u.v8[0] = *(const bf16x8*)(row + k0);
    u.v8[1] = *(const bf16x8*)(row + k0 + 16);
    return u.v16;
}

__device__ __forceinline__ f32x8 wmma_bf16(bf16x16 a, bf16x16 b, f32x8 c) {
    return __builtin_amdgcn_wmma_f32_16x16x32_bf16(
        /*neg_a=*/false, a, /*neg_b=*/false, b,
        /*c_mod=*/(short)0, c, /*reuse_a=*/false, /*reuse_b=*/false);
}

// ---------------------------------------------------------------------------
// Generic 128x128 output tile GEMM:  O = A[M,K] * W[K,N] + bias, K mult of 64.
// BK=64 stage depth: 16 WMMAs per wave per barrier pair. 256 threads = 8
// waves, wave grid 4(M) x 2(N): each wave owns 32x64 = 2x4 WMMA tiles.
// ---------------------------------------------------------------------------
template <typename AT, typename OT>
__device__ __forceinline__ void gemm_tile_128x128(
    const AT* __restrict__ A, int lda,
    const float* __restrict__ W, int ldw,
    const float* __restrict__ bias,
    OT* __restrict__ O, int ldo,
    int Ksize)
{
    __shared__ __attribute__((aligned(16))) __bf16 lA[128 * 64];
    __shared__ __attribute__((aligned(16))) __bf16 lB[128 * 64];

    const int t    = threadIdx.x;
    const int lane = t & 31;
    const int wid  = t >> 5;
    const int wm   = (wid & 3) * 32;
    const int wn   = (wid >> 2) * 64;
    const int rowM = blockIdx.y * 128;
    const int colN = blockIdx.x * 128;

    // staging coordinates
    const int sm  = t >> 1;            // A: row in tile (0..127)
    const int skb = (t & 1) * 32;      // A: k sub-chunk (0/32)
    const int sk  = t >> 2;            // W: k row in tile (0..63)
    const int snb = (t & 3) * 32;      // W: n sub-chunk (0/32/64/96)

    f32x8 acc[2][4];
    const f32x8 zero = {0.f, 0.f, 0.f, 0.f, 0.f, 0.f, 0.f, 0.f};
#pragma unroll
    for (int i = 0; i < 2; ++i)
#pragma unroll
        for (int j = 0; j < 4; ++j) acc[i][j] = zero;

    for (int kt = 0; kt < Ksize; kt += 64) {
        // prefetch next K tile into L2 while this one computes
        if (kt + 64 < Ksize) {
            __builtin_prefetch(A + (size_t)(rowM + sm) * lda + kt + 64 + skb, 0, 0);
            __builtin_prefetch(W + (size_t)(kt + 64 + sk) * ldw + colN + snb, 0, 0);
        }
        // stage A tile [128][64] -> bf16 (vectorized, contiguous)
        stage_chunk16(A + (size_t)(rowM + sm) * lda + kt + skb,      lA + sm * 64 + skb);
        stage_chunk16(A + (size_t)(rowM + sm) * lda + kt + skb + 16, lA + sm * 64 + skb + 16);
        {   // stage W tile [64][128] transposed -> lB[n][k]
            const float* src = W + (size_t)(kt + sk) * ldw + colN + snb;
            const f32x4* s4 = (const f32x4*)src;
#pragma unroll
            for (int q = 0; q < 8; ++q) {
                f32x4 vq = s4[q];
#pragma unroll
                for (int j = 0; j < 4; ++j)
                    lB[(snb + q * 4 + j) * 64 + sk] = (__bf16)vq[j];
            }
        }
        __syncthreads();

#pragma unroll
        for (int ks = 0; ks < 64; ks += 32) {
            bf16x16 af[2], bfr[4];
#pragma unroll
            for (int mi = 0; mi < 2; ++mi)
                af[mi] = load_a_frag(lA, 64, wm + mi * 16, ks, lane);
#pragma unroll
            for (int ni = 0; ni < 4; ++ni)
                bfr[ni] = load_b_frag(lB, 64, wn + ni * 16, ks, lane);
#pragma unroll
            for (int mi = 0; mi < 2; ++mi)
#pragma unroll
                for (int ni = 0; ni < 4; ++ni)
                    acc[mi][ni] = wmma_bf16(af[mi], bfr[ni], acc[mi][ni]);
        }
        __syncthreads();
    }

    const int half = lane >> 4;
    const int cn   = lane & 15;
#pragma unroll
    for (int mi = 0; mi < 2; ++mi)
#pragma unroll
        for (int ni = 0; ni < 4; ++ni) {
            int gn   = colN + wn + ni * 16 + cn;
            float bv = bias[gn];
#pragma unroll
            for (int r = 0; r < 8; ++r) {
                int gm = rowM + wm + mi * 16 + r + 8 * half;
                store_out(O + (size_t)gm * ldo + gn, acc[mi][ni][r] + bv);
            }
        }
}

// ------------------------- kernel 1: fused QKV projections ------------------
__global__ __launch_bounds__(256) void qkv_proj_kernel(
    const float* __restrict__ q, const float* __restrict__ k, const float* __restrict__ v,
    const float* __restrict__ Wq, const float* __restrict__ Wk, const float* __restrict__ Wv,
    const float* __restrict__ bq, const float* __restrict__ bk, const float* __restrict__ bv,
    __bf16* __restrict__ Qo, __bf16* __restrict__ Ko, __bf16* __restrict__ Vo)
{
    const float* A; const float* W; const float* bias; __bf16* O;
    if (blockIdx.z == 0)      { A = q; W = Wq; bias = bq; O = Qo; }
    else if (blockIdx.z == 1) { A = k; W = Wk; bias = bk; O = Ko; }
    else                      { A = v; W = Wv; bias = bv; O = Vo; }
    gemm_tile_128x128<float, __bf16>(A, Dc, W, Dc, bias, O, Dc, Dc);
}

// ------------------------- kernel 2: scaled scores Q Kh^T -------------------
// K = DK = 64: single 64-deep stage, one barrier, 16 WMMAs per wave.
__global__ __launch_bounds__(256) void scores_kernel(
    const __bf16* __restrict__ Qw, const __bf16* __restrict__ Kw,
    float* __restrict__ attn)
{
    const int bh = blockIdx.z;
    const int b  = bh >> 4;
    const int h  = bh & 15;
    const int rowM = blockIdx.y * 128;
    const int colN = blockIdx.x * 128;

    __shared__ __attribute__((aligned(16))) __bf16 lA[128 * 64];
    __shared__ __attribute__((aligned(16))) __bf16 lB[128 * 64];

    const int t = threadIdx.x, lane = t & 31, wid = t >> 5;
    const int wm = (wid & 3) * 32, wn = (wid >> 2) * 64;
    const int sm = t >> 1, skb = (t & 1) * 32;

    const __bf16* Qb = Qw + (size_t)b * Sc * Dc + (size_t)h * DKc;
    const __bf16* Kb = Kw + (size_t)b * Sc * Dc + (size_t)h * DKc;

    // A = Q-head rows (bf16, k-contiguous); Bt[n][k] = Kh rows (direct copy)
    stage_chunk16(Qb + (size_t)(rowM + sm) * Dc + skb,      lA + sm * 64 + skb);
    stage_chunk16(Qb + (size_t)(rowM + sm) * Dc + skb + 16, lA + sm * 64 + skb + 16);
    stage_chunk16(Kb + (size_t)(colN + sm) * Dc + skb,      lB + sm * 64 + skb);
    stage_chunk16(Kb + (size_t)(colN + sm) * Dc + skb + 16, lB + sm * 64 + skb + 16);
    __syncthreads();

    f32x8 acc[2][4];
    const f32x8 zero = {0.f, 0.f, 0.f, 0.f, 0.f, 0.f, 0.f, 0.f};
#pragma unroll
    for (int i = 0; i < 2; ++i)
#pragma unroll
        for (int j = 0; j < 4; ++j) acc[i][j] = zero;

#pragma unroll
    for (int ks = 0; ks < 64; ks += 32) {
        bf16x16 af[2], bfr[4];
#pragma unroll
        for (int mi = 0; mi < 2; ++mi)
            af[mi] = load_a_frag(lA, 64, wm + mi * 16, ks, lane);
#pragma unroll
        for (int ni = 0; ni < 4; ++ni)
            bfr[ni] = load_b_frag(lB, 64, wn + ni * 16, ks, lane);
#pragma unroll
        for (int mi = 0; mi < 2; ++mi)
#pragma unroll
            for (int ni = 0; ni < 4; ++ni)
                acc[mi][ni] = wmma_bf16(af[mi], bfr[ni], acc[mi][ni]);
    }

    const int half = lane >> 4, cn = lane & 15;
    const float scale = 0.125f;   // 1/sqrt(64)
#pragma unroll
    for (int mi = 0; mi < 2; ++mi)
#pragma unroll
        for (int ni = 0; ni < 4; ++ni) {
            int gn = colN + wn + ni * 16 + cn;
#pragma unroll
            for (int r = 0; r < 8; ++r) {
                int gm = rowM + wm + mi * 16 + r + 8 * half;
                attn[((size_t)bh * Sc + gm) * Sc + gn] = acc[mi][ni][r] * scale;
            }
        }
}

// ------------------------- kernel 3: row softmax in place -------------------
__global__ __launch_bounds__(256) void softmax_kernel(float* __restrict__ attn)
{
    const int t = threadIdx.x;
    float* row = attn + (size_t)blockIdx.x * Sc;
    __shared__ float red[256];

    // each thread owns 8 contiguous columns -> two b128 loads
    f32x4* rv = (f32x4*)(row + t * 8);
    f32x4 v0 = rv[0], v1 = rv[1];

    float m = v0[0];
#pragma unroll
    for (int i = 1; i < 4; ++i) m = fmaxf(m, v0[i]);
#pragma unroll
    for (int i = 0; i < 4; ++i) m = fmaxf(m, v1[i]);
    red[t] = m;
    __syncthreads();
    for (int s = 128; s > 0; s >>= 1) {
        if (t < s) red[t] = fmaxf(red[t], red[t + s]);
        __syncthreads();
    }
    const float rmax = red[0];
    __syncthreads();

    float sum = 0.f;
#pragma unroll
    for (int i = 0; i < 4; ++i) { v0[i] = __expf(v0[i] - rmax); sum += v0[i]; }
#pragma unroll
    for (int i = 0; i < 4; ++i) { v1[i] = __expf(v1[i] - rmax); sum += v1[i]; }
    red[t] = sum;
    __syncthreads();
    for (int s = 128; s > 0; s >>= 1) {
        if (t < s) red[t] += red[t + s];
        __syncthreads();
    }
    const float inv = 1.0f / red[0];
    rv[0] = v0 * inv;
    rv[1] = v1 * inv;
}

// ------------------------- kernel 4: X = attn @ V-head ----------------------
__global__ __launch_bounds__(256) void pv_kernel(
    const float* __restrict__ attn, const __bf16* __restrict__ Vw,
    __bf16* __restrict__ Xo)
{
    const int bh = blockIdx.z;
    const int b  = bh >> 4;
    const int h  = bh & 15;
    const int rowM = blockIdx.y * 128;     // N tile = full DK (64)

    __shared__ __attribute__((aligned(16))) __bf16 lA[128 * 64];
    __shared__ __attribute__((aligned(16))) __bf16 lB[64 * 64];

    const int t = threadIdx.x, lane = t & 31, wid = t >> 5;
    const int wm = (wid & 3) * 32, wn = (wid >> 2) * 32;
    const int sm = t >> 1, skb = (t & 1) * 32;   // A staging
    const int sk = t >> 2, snb = (t & 3) * 16;   // V staging (k row, n chunk)

    f32x8 acc[2][2];
    const f32x8 zero = {0.f, 0.f, 0.f, 0.f, 0.f, 0.f, 0.f, 0.f};
#pragma unroll
    for (int i = 0; i < 2; ++i)
#pragma unroll
        for (int j = 0; j < 2; ++j) acc[i][j] = zero;

    const float*  Ab = attn + (size_t)bh * Sc * Sc;
    const __bf16* Vb = Vw + (size_t)b * Sc * Dc + (size_t)h * DKc;

    for (int kt = 0; kt < Sc; kt += 64) {
        if (kt + 64 < Sc) {
            __builtin_prefetch(Ab + (size_t)(rowM + sm) * Sc + kt + 64 + skb, 0, 0);
            __builtin_prefetch(Vb + (size_t)(kt + 64 + sk) * Dc + snb, 0, 0);
        }
        // A = attn tile [128][64], f32 -> bf16 (vectorized)
        stage_chunk16(Ab + (size_t)(rowM + sm) * Sc + kt + skb,      lA + sm * 64 + skb);
        stage_chunk16(Ab + (size_t)(rowM + sm) * Sc + kt + skb + 16, lA + sm * 64 + skb + 16);
        {   // Bt[n][k] = Vh[kt+k][n] (transpose on store)
            const bf16x8* vs = (const bf16x8*)(Vb + (size_t)(kt + sk) * Dc + snb);
            bf16x8 v0 = vs[0], v1 = vs[1];
#pragma unroll
            for (int j = 0; j < 8; ++j) {
                lB[(snb + j) * 64 + sk]     = v0[j];
                lB[(snb + 8 + j) * 64 + sk] = v1[j];
            }
        }
        __syncthreads();

#pragma unroll
        for (int ks = 0; ks < 64; ks += 32) {
            bf16x16 af[2], bfr[2];
#pragma unroll
            for (int mi = 0; mi < 2; ++mi)
                af[mi] = load_a_frag(lA, 64, wm + mi * 16, ks, lane);
#pragma unroll
            for (int ni = 0; ni < 2; ++ni)
                bfr[ni] = load_b_frag(lB, 64, wn + ni * 16, ks, lane);
#pragma unroll
            for (int mi = 0; mi < 2; ++mi)
#pragma unroll
                for (int ni = 0; ni < 2; ++ni)
                    acc[mi][ni] = wmma_bf16(af[mi], bfr[ni], acc[mi][ni]);
        }
        __syncthreads();
    }

    const int half = lane >> 4, cn = lane & 15;
#pragma unroll
    for (int mi = 0; mi < 2; ++mi)
#pragma unroll
        for (int ni = 0; ni < 2; ++ni) {
            int gn = wn + ni * 16 + cn;
#pragma unroll
            for (int r = 0; r < 8; ++r) {
                int gm = rowM + wm + mi * 16 + r + 8 * half;
                Xo[((size_t)b * Sc + gm) * Dc + (size_t)h * DKc + gn] =
                    to_bf16(acc[mi][ni][r]);
            }
        }
}

// ------------------------- kernel 5: output projection ----------------------
__global__ __launch_bounds__(256) void out_proj_kernel(
    const __bf16* __restrict__ X, const float* __restrict__ Wo,
    const float* __restrict__ bo, float* __restrict__ out)
{
    gemm_tile_128x128<__bf16, float>(X, Dc, Wo, Dc, bo, out, Dc, Dc);
}

// ---------------------------------------------------------------------------
extern "C" void kernel_launch(void* const* d_in, const int* in_sizes, int n_in,
                              void* d_out, int out_size, void* d_ws, size_t ws_size,
                              hipStream_t stream)
{
    const float* q  = (const float*)d_in[0];
    const float* k  = (const float*)d_in[1];
    const float* v  = (const float*)d_in[2];
    const float* Wq = (const float*)d_in[3];
    const float* bq = (const float*)d_in[4];
    const float* Wk = (const float*)d_in[5];
    const float* bk = (const float*)d_in[6];
    const float* Wv = (const float*)d_in[7];
    const float* bv = (const float*)d_in[8];
    const float* Wo = (const float*)d_in[9];
    const float* bo = (const float*)d_in[10];

    float* out  = (float*)d_out;
    float* attn = out + (size_t)Bc * Sc * Dc;

    __bf16* wsQ = (__bf16*)d_ws;
    __bf16* wsK = wsQ + (size_t)Bc * Sc * Dc;
    __bf16* wsV = wsK + (size_t)Bc * Sc * Dc;
    __bf16* wsX = wsV + (size_t)Bc * Sc * Dc;

    dim3 blk(256);
    qkv_proj_kernel<<<dim3(Dc / 128, Bc * Sc / 128, 3), blk, 0, stream>>>(
        q, k, v, Wq, Wk, Wv, bq, bk, bv, wsQ, wsK, wsV);
    scores_kernel<<<dim3(Sc / 128, Sc / 128, Bc * Hc), blk, 0, stream>>>(
        wsQ, wsK, attn);
    softmax_kernel<<<dim3(Bc * Hc * Sc), blk, 0, stream>>>(attn);
    pv_kernel<<<dim3(1, Sc / 128, Bc * Hc), blk, 0, stream>>>(attn, wsV, wsX);
    out_proj_kernel<<<dim3(Dc / 128, Bc * Sc / 128, 1), blk, 0, stream>>>(
        wsX, Wo, bo, out);
}
